// GINEncoder_68848325755451
// MI455X (gfx1250) — matrix operand order repaired and checked by reading, not compiled, not run on hardware
//
#include <hip/hip_runtime.h>

// ---------------------------------------------------------------------------
// CDNA5 (gfx1250) fused GINE encoder.
// wave32, WMMA bf16 16x16x32 with f32 accumulate.
// ---------------------------------------------------------------------------

typedef __attribute__((ext_vector_type(16))) __bf16 v16bf;
typedef __attribute__((ext_vector_type(8)))  float  v8f;

#define WMMA_BF16(A, B, C) \
  __builtin_amdgcn_wmma_f32_16x16x32_bf16(false, (A), false, (B), (short)0, (C), false, false)

static __device__ __forceinline__ v8f vzero8() {
  v8f z;
#pragma unroll
  for (int i = 0; i < 8; ++i) z[i] = 0.0f;
  return z;
}

// A-fragment (16x32 bf16, ISA layout) from f32 row-major memory.
// p = &row[lane&15] data, already offset by kb = (lane>=16 ? 8 : 0).
// elements j   -> K = kb + j        (j = 0..7)
// elements j+8 -> K = kb + 16 + j
static __device__ __forceinline__ v16bf load_a_f32(const float* __restrict__ p) {
  v16bf a;
#pragma unroll
  for (int j = 0; j < 8; ++j) {
    a[j]     = (__bf16)p[j];
    a[j + 8] = (__bf16)p[16 + j];
  }
  return a;
}

static __device__ __forceinline__ v16bf load_a_bf16(const __bf16* __restrict__ p) {
  v16bf a;
#pragma unroll
  for (int j = 0; j < 8; ++j) {
    a[j]     = p[j];
    a[j + 8] = p[16 + j];
  }
  return a;
}

// ---------------------------------------------------------------------------
// Fused edge kernel:  agg[dst] += relu( x[src] + edge_attr @ we + be )
// One wave per 16-edge tile; we (32x128) kept in registers as 8 B-fragments.
// Indices read as low-dword of int64 (node ids < 2^31); per-row byte offsets
// hoisted so the 8 column tiles use constant instruction immediate offsets.
// Gathers for a row are batched ahead of the atomic sweep so their latency
// pipelines instead of serializing load->wait->atomic per element.
// ---------------------------------------------------------------------------
__global__ __launch_bounds__(256) void gine_edge_kernel(
    const float* __restrict__ x, const int* __restrict__ src32,
    const int* __restrict__ dst32, const float* __restrict__ eattr,
    const float* __restrict__ we, const float* __restrict__ be,
    float* __restrict__ agg, int E) {
  const int lane  = threadIdx.x & 31;
  const int wid   = threadIdx.x >> 5;
  const int gwave = blockIdx.x * (blockDim.x >> 5) + wid;
  const int nwave = gridDim.x * (blockDim.x >> 5);

  const int hlf   = lane >> 4;      // 0: lanes 0-15, 1: lanes 16-31
  const int l16   = lane & 15;
  const int kb    = hlf * 8;        // A K-base within 32
  const int rbase = hlf * 8;        // C row base

  // B fragments for we[32,128] (K x N, row-major, ld=128), 8 column tiles.
  // element j -> K=hlf*8+(j&7)+(j>>3)*16, col=lane&15.
  v16bf B[8];
  float bias[8];
#pragma unroll
  for (int t = 0; t < 8; ++t) {
    const int col = t * 16 + l16;
#pragma unroll
    for (int j = 0; j < 16; ++j)
      B[t][j] = (__bf16)we[(size_t)(hlf * 8 + (j & 7) + (j >> 3) * 16) * 128 + col];
    bias[t] = be[col];
  }

  const char* __restrict__ xb = (const char*)x;
  char* __restrict__ ab       = (char*)agg;

  const int ntiles = E >> 4;
  for (int tile = gwave; tile < ntiles; tile += nwave) {
    const int e0 = tile << 4;

    // A fragment: 16 edge_attr rows x K=32
    const v16bf A = load_a_f32(eattr + (size_t)(e0 + l16) * 32 + kb);

    // all 8 column-tile WMMAs up front; C stays in registers
    v8f c[8];
#pragma unroll
    for (int t = 0; t < 8; ++t) {
      c[t] = vzero8();
      c[t] = WMMA_BF16(A, B[t], c[t]);
    }

    // per-row gather/scatter: one address per row, col tiles via imm offsets
#pragma unroll
    for (int v = 0; v < 8; ++v) {
      const int e = e0 + rbase + v;
      const unsigned soff = ((unsigned)src32[2 * e]) << 9;  // node * 512B
      const unsigned doff = ((unsigned)dst32[2 * e]) << 9;
      const float* xp = (const float*)(xb + soff) + l16;
      float*       ap = (float*)(ab + doff) + l16;

      // batch the 8 gathers so they pipeline
      float xv[8];
#pragma unroll
      for (int t = 0; t < 8; ++t) xv[t] = xp[t * 16];  // +t*64B immediate

#pragma unroll
      for (int t = 0; t < 8; ++t) {
        float m = c[t][v] + bias[t] + xv[t];
        m = m > 0.0f ? m : 0.0f;
        atomicAdd(&ap[t * 16], m);                     // +t*64B immediate
      }
    }
  }
}

// scalar tail (only used when E % 16 != 0)
__global__ void gine_edge_tail_kernel(
    const float* __restrict__ x, const int* __restrict__ src32,
    const int* __restrict__ dst32, const float* __restrict__ eattr,
    const float* __restrict__ we, const float* __restrict__ be,
    float* __restrict__ agg, int e_start, int E) {
  const int e = e_start + blockIdx.x;
  if (e >= E) return;
  const int col = threadIdx.x;  // 128 threads
  const float* a = eattr + (size_t)e * 32;
  float acc = be[col];
  for (int k = 0; k < 32; ++k) acc += a[k] * we[(size_t)k * 128 + col];
  float m = acc + x[(size_t)(unsigned)src32[2 * e] * 128 + col];
  m = m > 0.0f ? m : 0.0f;
  atomicAdd(&agg[(size_t)(unsigned)dst32[2 * e] * 128 + col], m);
}

// ---------------------------------------------------------------------------
// Node MLP:  out = act2( relu(in @ W1 + b1) @ W2 + b2 )
// Weights staged in LDS as bf16 fragments; intermediate tile round-trips
// through a per-wave LDS buffer (LDS is in-order per wave -> no barrier).
// ---------------------------------------------------------------------------
__global__ __launch_bounds__(256) void gine_mlp_kernel(
    const float* __restrict__ in, const float* __restrict__ W1,
    const float* __restrict__ b1, const float* __restrict__ W2,
    const float* __restrict__ b2, float* __restrict__ out, int nrows,
    int final_relu) {
  // fragment-major bf16 weight staging: frag = kc*8 + t, [frag][lane][j]
  __shared__ __align__(32) __bf16 w1s[32][32][16];
  __shared__ __align__(32) __bf16 w2s[32][32][16];
  __shared__ __align__(32) __bf16 hinter[8][16 * 128];  // per-wave 16x128 tile

  for (int idx = threadIdx.x; idx < 32 * 32; idx += blockDim.x) {
    const int frag = idx >> 5, ln = idx & 31;
    const int kc = frag >> 3, t = frag & 7;
    const int colb = t * 16 + (ln & 15);
    const int rowb = kc * 32 + (ln >> 4) * 8;
#pragma unroll
    for (int j = 0; j < 16; ++j) {
      w1s[frag][ln][j] = (__bf16)W1[(size_t)(rowb + (j & 7) + (j >> 3) * 16) * 128 + colb];
      w2s[frag][ln][j] = (__bf16)W2[(size_t)(rowb + (j & 7) + (j >> 3) * 16) * 128 + colb];
    }
  }
  __syncthreads();

  const int lane  = threadIdx.x & 31;
  const int wid   = threadIdx.x >> 5;
  const int gwave = blockIdx.x * (blockDim.x >> 5) + wid;
  const int nwave = gridDim.x * (blockDim.x >> 5);
  const int hlf   = lane >> 4;
  const int l16   = lane & 15;
  const int rbase = hlf * 8;
  __bf16* hb = &hinter[wid][0];

  const int ntiles = nrows >> 4;
  for (int tile = gwave; tile < ntiles; tile += nwave) {
    const size_t r0 = (size_t)tile << 4;

    // ---- GEMM1: relu(in @ W1 + b1) -> LDS tile (bf16) ----
    v8f c[8];
#pragma unroll
    for (int t = 0; t < 8; ++t) c[t] = vzero8();
#pragma unroll
    for (int kc = 0; kc < 4; ++kc) {
      const v16bf A = load_a_f32(in + (r0 + l16) * 128 + kc * 32 + hlf * 8);
#pragma unroll
      for (int t = 0; t < 8; ++t) {
        const v16bf B = *(const v16bf*)&w1s[kc * 8 + t][lane][0];
        c[t] = WMMA_BF16(A, B, c[t]);
      }
    }
#pragma unroll
    for (int t = 0; t < 8; ++t) {
      const int col = t * 16 + l16;
      const float bb = b1[col];
#pragma unroll
      for (int v = 0; v < 8; ++v) {
        float h = c[t][v] + bb;
        h = h > 0.0f ? h : 0.0f;
        hb[(rbase + v) * 128 + col] = (__bf16)h;
      }
    }
    __asm__ __volatile__("" ::: "memory");  // keep LDS store->load order

    // ---- GEMM2: tile @ W2 + b2 ----
#pragma unroll
    for (int t = 0; t < 8; ++t) c[t] = vzero8();
#pragma unroll
    for (int kc = 0; kc < 4; ++kc) {
      const v16bf A = load_a_bf16(hb + (size_t)l16 * 128 + kc * 32 + hlf * 8);
#pragma unroll
      for (int t = 0; t < 8; ++t) {
        const v16bf B = *(const v16bf*)&w2s[kc * 8 + t][lane][0];
        c[t] = WMMA_BF16(A, B, c[t]);
      }
    }
#pragma unroll
    for (int t = 0; t < 8; ++t) {
      const int col = t * 16 + l16;
      const float bb = b2[col];
#pragma unroll
      for (int v = 0; v < 8; ++v) {
        float h = c[t][v] + bb;
        if (final_relu) h = h > 0.0f ? h : 0.0f;
        out[(r0 + rbase + v) * 128 + col] = h;
      }
    }
    __asm__ __volatile__("" ::: "memory");  // WAR on hb across iterations
  }
}

// scalar tail (only used when nrows % 16 != 0)
__global__ void gine_mlp_tail_kernel(
    const float* __restrict__ in, const float* __restrict__ W1,
    const float* __restrict__ b1, const float* __restrict__ W2,
    const float* __restrict__ b2, float* __restrict__ out, int r_start,
    int nrows, int final_relu) {
  const int r = r_start + blockIdx.x;
  if (r >= nrows) return;
  const int col = threadIdx.x;  // 128 threads
  __shared__ float h1[128];
  const float* xr = in + (size_t)r * 128;
  float acc = b1[col];
  for (int k = 0; k < 128; ++k) acc += xr[k] * W1[(size_t)k * 128 + col];
  h1[col] = acc > 0.0f ? acc : 0.0f;
  __syncthreads();
  float acc2 = b2[col];
  for (int k = 0; k < 128; ++k) acc2 += h1[k] * W2[(size_t)k * 128 + col];
  if (final_relu) acc2 = acc2 > 0.0f ? acc2 : 0.0f;
  out[(size_t)r * 128 + col] = acc2;
}

// agg := src  (vectorized copy; also serves as the "+x" self term, eps=0)
__global__ void copy_f4_kernel(const float4* __restrict__ s,
                               float4* __restrict__ d, long long n4) {
  long long i = (long long)blockIdx.x * blockDim.x + threadIdx.x;
  const long long stride = (long long)gridDim.x * blockDim.x;
  for (; i < n4; i += stride) d[i] = s[i];
}

// ---------------------------------------------------------------------------
extern "C" void kernel_launch(void* const* d_in, const int* in_sizes, int n_in,
                              void* d_out, int out_size, void* d_ws,
                              size_t ws_size, hipStream_t stream) {
  const float* x     = (const float*)d_in[0];
  const int*   eidx  = (const int*)d_in[1];   // int64 data, low dwords used
  const float* eattr = (const float*)d_in[2];
  const float *w1_0 = (const float*)d_in[3],  *b1_0 = (const float*)d_in[4];
  const float *w2_0 = (const float*)d_in[5],  *b2_0 = (const float*)d_in[6];
  const float *we_0 = (const float*)d_in[7],  *be_0 = (const float*)d_in[8];
  const float *w1_1 = (const float*)d_in[9],  *b1_1 = (const float*)d_in[10];
  const float *w2_1 = (const float*)d_in[11], *b2_1 = (const float*)d_in[12];
  const float *we_1 = (const float*)d_in[13], *be_1 = (const float*)d_in[14];
  const float *fc1w = (const float*)d_in[15], *fc1b = (const float*)d_in[16];
  const float *fc2w = (const float*)d_in[17], *fc2b = (const float*)d_in[18];

  const int N = in_sizes[0] / 128;
  const int E = in_sizes[2] / 32;
  const long long NF  = (long long)N * 128;
  const long long NF4 = NF / 4;

  float* agg = (float*)d_ws;        // [N,128]
  float* hbf = agg + NF;            // [N,128]
  const int* srcp = eidx;           // edge_index[0], int64 -> stride-2 dwords
  const int* dstp = eidx + 2 * E;   // edge_index[1]

  const dim3 blk(256);
  const dim3 grid_copy(512), grid_edge(1024), grid_mlp(256);
  const int edge_main = E & ~15, remE = E - edge_main;
  const int row_main = N & ~15, remN = N - row_main;

  // ---------------- layer 0 ----------------
  copy_f4_kernel<<<grid_copy, blk, 0, stream>>>((const float4*)x, (float4*)agg, NF4);
  gine_edge_kernel<<<grid_edge, blk, 0, stream>>>(x, srcp, dstp, eattr, we_0, be_0, agg, E);
  if (remE)
    gine_edge_tail_kernel<<<dim3(remE), dim3(128), 0, stream>>>(
        x, srcp, dstp, eattr, we_0, be_0, agg, edge_main, E);
  gine_mlp_kernel<<<grid_mlp, blk, 0, stream>>>(agg, w1_0, b1_0, w2_0, b2_0, hbf, N, 1);
  if (remN)
    gine_mlp_tail_kernel<<<dim3(remN), dim3(128), 0, stream>>>(
        agg, w1_0, b1_0, w2_0, b2_0, hbf, row_main, N, 1);

  // ---------------- layer 1 ----------------
  copy_f4_kernel<<<grid_copy, blk, 0, stream>>>((const float4*)hbf, (float4*)agg, NF4);
  gine_edge_kernel<<<grid_edge, blk, 0, stream>>>(hbf, srcp, dstp, eattr, we_1, be_1, agg, E);
  if (remE)
    gine_edge_tail_kernel<<<dim3(remE), dim3(128), 0, stream>>>(
        hbf, srcp, dstp, eattr, we_1, be_1, agg, edge_main, E);
  gine_mlp_kernel<<<grid_mlp, blk, 0, stream>>>(agg, w1_1, b1_1, w2_1, b2_1, hbf, N, 1);
  if (remN)
    gine_mlp_tail_kernel<<<dim3(remN), dim3(128), 0, stream>>>(
        agg, w1_1, b1_1, w2_1, b2_1, hbf, row_main, N, 1);

  // ---------------- head ----------------
  gine_mlp_kernel<<<grid_mlp, blk, 0, stream>>>(hbf, fc1w, fc1b, fc2w, fc2b,
                                                (float*)d_out, N, 0);
  if (remN)
    gine_mlp_tail_kernel<<<dim3(remN), dim3(128), 0, stream>>>(
        hbf, fc1w, fc1b, fc2w, fc2b, (float*)d_out, row_main, N, 0);
}
